// LSTMEDModule_2095944041061
// MI455X (gfx1250) — matrix operand order repaired and checked by reading, not compiled, not run on hardware
//
#include <hip/hip_runtime.h>
#include <stdint.h>

// ---------------------------------------------------------------------------
// LSTM encoder-decoder for MI455X (gfx1250, wave32, WMMA).
// B=128,T=2048,F=64,H=256. Batch-partitioned persistent recurrence:
//   grid = 8 blocks x 256 threads (8 waves); block b owns batch rows 16b..16b+15.
//   h state: LDS bf16 [16][256]; c state: registers (WMMA C-layout).
//   Per step per wave: 10 K-slices x 8 N-tiles = 80 v_wmma_f32_16x16x32_bf16.
// Weights pre-packed (one-time kernels) into WMMA B-operand lane layout, bf16,
// streamed from L2 each step. A laundered zero *offset* (not pointer) blocks
// LICM from hoisting the loop-invariant fragment loads (round-1 spill storm)
// while preserving global address-space provenance so the backend emits
// global_load_b128 (round-2 regression was flat_load_b128 coupling DScnt).
// ---------------------------------------------------------------------------

typedef __attribute__((ext_vector_type(16))) __bf16        v16bf;
typedef __attribute__((ext_vector_type(8)))  float         v8f;
typedef __attribute__((ext_vector_type(4)))  unsigned int  u32x4;

constexpr int BB = 128, TT = 2048, FF = 64, HH = 256, GG = 4 * HH;

union FragU { v16bf v; u32x4 q[2]; };

__device__ __forceinline__ unsigned short f2bf(float f) {
  union { float f; unsigned u; } v; v.f = f;
  unsigned r = v.u + 0x7FFFu + ((v.u >> 16) & 1u);   // round-to-nearest-even
  return (unsigned short)(r >> 16);
}

// B-operand fragment from global (pre-swizzled): lane base = 16 bf16, 32B.
__device__ __forceinline__ v16bf ldfrag_g(const unsigned short* p) {
  FragU f;
  f.q[0] = *reinterpret_cast<const u32x4*>(p);
  f.q[1] = *reinterpret_cast<const u32x4*>(p + 8);
  return f.v;
}

// A-operand fragment from row-major bf16 LDS tile (ISA 16-bit A layout):
// lanes 0-15 hold K {kb+0..7, kb+16..23}; lanes 16-31 hold {kb+8..15, kb+24..31}.
__device__ __forceinline__ v16bf ldfragA(const unsigned short* row, int kb, int half) {
  FragU f;
  f.q[0] = *reinterpret_cast<const u32x4*>(row + kb + half * 8);
  f.q[1] = *reinterpret_cast<const u32x4*>(row + kb + 16 + half * 8);
  return f.v;
}

__device__ __forceinline__ v8f wmma_bf16(v16bf a, v16bf b, v8f c) {
  return __builtin_amdgcn_wmma_f32_16x16x32_bf16(false, a, false, b, (short)0, c,
                                                 false, false);
}

__device__ __forceinline__ float sigm(float x) { return 1.0f / (1.0f + __expf(-x)); }
__device__ __forceinline__ float tanh_f(float x) {
  x = fminf(fmaxf(x, -15.f), 15.f);
  float e = __expf(2.0f * x);
  return (e - 1.0f) / (e + 1.0f);
}

// --------------------------- prep kernels ----------------------------------

__global__ void k_cvt_bf16(const float* __restrict__ in,
                           unsigned short* __restrict__ o, int n) {
  int i = blockIdx.x * blockDim.x + threadIdx.x;
  int stride = gridDim.x * blockDim.x;
  for (; i < n; i += stride) o[i] = f2bf(in[i]);
}

// Pack W[N][K] (row-major f32, gates = act @ W^T so B[k][n] = W[n][k]) into
// P[jn][kt][lane][16] bf16 matching the WMMA B-operand lane layout:
//   lane l holds column n = jn*16 + (l&15), K = kt*32 + (l>>4)*8 + (i&7) + (i>>3)*16.
__global__ void k_pack(const float* __restrict__ W,
                       unsigned short* __restrict__ P, int N, int K) {
  int idx = blockIdx.x * blockDim.x + threadIdx.x;
  if (idx >= N * K) return;
  int i    = idx & 15;
  int lane = (idx >> 4) & 31;
  int frag = idx >> 9;
  int KT = K >> 5;
  int kt = frag % KT;
  int jn = frag / KT;
  int n = jn * 16 + (lane & 15);
  int k = kt * 32 + (lane >> 4) * 8 + (i & 7) + (i >> 3) * 16;
  P[idx] = f2bf(W[n * K + k]);
}

// --------------------------- main recurrence -------------------------------

__global__ __launch_bounds__(256, 1)
void lstm_ed_main(const unsigned short* __restrict__ xbf,    // [B][T][F] bf16
                  const unsigned short* __restrict__ wihE,   // packed [64][2][512]
                  const unsigned short* __restrict__ whhE,   // packed [64][8][512]
                  const unsigned short* __restrict__ wihD,
                  const unsigned short* __restrict__ whhD,
                  const unsigned short* __restrict__ woutP,  // packed [4][8][512]
                  const float* __restrict__ bE,              // [1024]
                  const float* __restrict__ bD,              // [1024]
                  const float* __restrict__ bout,            // [64]
                  float* __restrict__ out)                   // [B][T][F]
{
  __shared__ unsigned short lds_x[16 * FF];   // x[t] tile, bf16
  __shared__ unsigned short lds_h[16 * HH];   // h state, bf16

  const int tid  = threadIdx.x;
  const int wave = tid >> 5;
  const int lane = tid & 31;
  const int lrow = lane & 15;
  const int half = lane >> 4;
  const int b0   = blockIdx.x * 16;

  for (int i = tid; i < 16 * HH; i += 256) lds_h[i] = 0;

  // c state in WMMA C-layout registers: q=0 -> h-column tile `wave`,
  // q=1 -> tile `wave+8`; element r is row m = half*8 + r, col = tile*16+lrow.
  float cst[2][8];
  #pragma unroll
  for (int q = 0; q < 2; ++q)
    #pragma unroll
    for (int r = 0; r < 8; ++r) cst[q][r] = 0.f;

  // Wave `wave` owns gate N-tiles jn = wave + 8p; p in {0,1}=i, {2,3}=f,
  // {4,5}=g, {6,7}=o with matched column offsets.
  float biasE[8], biasD[8];
  #pragma unroll
  for (int p = 0; p < 8; ++p) {
    int n = (wave + 8 * p) * 16 + lrow;
    biasE[p] = bE[n];
    biasD[p] = bD[n];
  }
  const float bo = (wave < 4) ? bout[wave * 16 + lrow] : 0.f;

  // cooperative x staging: thread -> (row, 4 cols)
  const int xr = tid >> 4;
  const int xc = (tid & 15) * 4;
  const unsigned short* xrow = xbf + (size_t)(b0 + xr) * TT * FF + xc;

  __syncthreads();

  #pragma unroll
  for (int phase = 0; phase < 2; ++phase) {          // 0=encoder fwd, 1=decoder bwd
    const unsigned short* wihP = phase ? wihD : wihE;
    const unsigned short* whhP = phase ? whhD : whhE;

    for (int s = 0; s < TT; ++s) {
      const int t = phase ? (TT - 1 - s) : s;

      // Opaque zero element-offset: defeats LICM on the weight-fragment loads
      // (addresses look per-iteration) while keeping the pointers derived from
      // kernel args -> addrspace(1) -> global_load_b128, not flat_load_b128.
      unsigned zofs = 0;
      asm volatile("" : "+s"(zofs));
      const unsigned short* wih = wihP + zofs;
      const unsigned short* whh = whhP + zofs;
      const unsigned short* wout = woutP + zofs;

      // stage x[:, t, :] tile into LDS (8 bytes per thread)
      *reinterpret_cast<unsigned long long*>(&lds_x[xr * FF + xc]) =
          *reinterpret_cast<const unsigned long long*>(xrow + (size_t)t * FF);

      __syncthreads();  // x[t] visible; lds_h stable (pre-update h)

      // Decoder: out[:, t] = h @ Wout^T + bout  (h BEFORE cell update)
      if (phase && wave < 4) {
        v8f oa;
        #pragma unroll
        for (int r = 0; r < 8; ++r) oa[r] = bo;
        const unsigned short* wb = wout + (size_t)(wave * 8) * 512 + lane * 16;
        v16bf bf = ldfrag_g(wb);
        #pragma unroll
        for (int kt = 0; kt < 8; ++kt) {
          v16bf bn = (kt < 7) ? ldfrag_g(wb + (size_t)(kt + 1) * 512) : bf;
          v16bf a = ldfragA(&lds_h[lrow * HH], kt * 32, half);
          oa = wmma_bf16(a, bf, oa);
          bf = bn;
        }
        const int n = wave * 16 + lrow;
        float* op = out + (size_t)t * FF + n;
        #pragma unroll
        for (int r = 0; r < 8; ++r)
          op[(size_t)(b0 + half * 8 + r) * TT * FF] = oa[r];
      }

      // gates[16,1024] = x @ Wih^T + h @ Whh^T + b   (bias-initialized acc)
      v8f acc[8];
      #pragma unroll
      for (int p = 0; p < 8; ++p) {
        float bv = phase ? biasD[p] : biasE[p];
        #pragma unroll
        for (int r = 0; r < 8; ++r) acc[p][r] = bv;
      }
      #pragma unroll
      for (int kt = 0; kt < 2; ++kt) {               // K over F=64
        v16bf a = ldfragA(&lds_x[lrow * FF], kt * 32, half);
        const unsigned short* wb = wih + (size_t)kt * 512 + lane * 16;
        v16bf bf = ldfrag_g(wb + (size_t)(wave * 2) * 512);
        #pragma unroll
        for (int p = 0; p < 8; ++p) {
          int jn_next = wave + 8 * (p + 1);
          v16bf bn = (p < 7) ? ldfrag_g(wb + (size_t)(jn_next * 2) * 512) : bf;
          acc[p] = wmma_bf16(a, bf, acc[p]);
          bf = bn;
        }
      }
      #pragma unroll
      for (int kt = 0; kt < 8; ++kt) {               // K over H=256
        v16bf a = ldfragA(&lds_h[lrow * HH], kt * 32, half);
        const unsigned short* wb = whh + (size_t)kt * 512 + lane * 16;
        v16bf bf = ldfrag_g(wb + (size_t)(wave * 8) * 512);
        #pragma unroll
        for (int p = 0; p < 8; ++p) {
          int jn_next = wave + 8 * (p + 1);
          v16bf bn = (p < 7) ? ldfrag_g(wb + (size_t)(jn_next * 8) * 512) : bf;
          acc[p] = wmma_bf16(a, bf, acc[p]);
          bf = bn;
        }
      }

      // LSTM cell elementwise (PyTorch gate order i,f,g,o)
      unsigned short hnew[2][8];
      #pragma unroll
      for (int q = 0; q < 2; ++q) {
        #pragma unroll
        for (int r = 0; r < 8; ++r) {
          float iv = sigm(acc[q][r]);
          float fv = sigm(acc[q + 2][r]);
          float gv = tanh_f(acc[q + 4][r]);
          float ov = sigm(acc[q + 6][r]);
          float cn = fv * cst[q][r] + iv * gv;
          cst[q][r] = cn;
          hnew[q][r] = f2bf(ov * tanh_f(cn));
        }
      }

      __syncthreads();  // all LDS reads of this step complete

      // scatter h_new back to row-major LDS per C/D VGPR layout
      #pragma unroll
      for (int q = 0; q < 2; ++q) {
        const int col = (wave + 8 * q) * 16 + lrow;
        #pragma unroll
        for (int r = 0; r < 8; ++r)
          lds_h[(half * 8 + r) * HH + col] = hnew[q][r];
      }
      // next iteration's first __syncthreads publishes h_new
    }
  }
}

// --------------------------- host launcher ---------------------------------

extern "C" void kernel_launch(void* const* d_in, const int* in_sizes, int n_in,
                              void* d_out, int out_size, void* d_ws, size_t ws_size,
                              hipStream_t stream) {
  (void)in_sizes; (void)n_in; (void)out_size; (void)ws_size;

  const float* ts   = (const float*)d_in[0];
  const float* WihE = (const float*)d_in[1];
  const float* WhhE = (const float*)d_in[2];
  const float* bE   = (const float*)d_in[3];
  const float* WihD = (const float*)d_in[4];
  const float* WhhD = (const float*)d_in[5];
  const float* bD   = (const float*)d_in[6];
  const float* Wout = (const float*)d_in[7];
  const float* bout = (const float*)d_in[8];
  float* out = (float*)d_out;

  // workspace layout (ushort units), all 32B aligned: ~35 MB total
  unsigned short* ws = (unsigned short*)d_ws;
  size_t off = 0;
  unsigned short* xbf   = ws + off; off += (size_t)BB * TT * FF;  // 33.5 MB
  unsigned short* wihEp = ws + off; off += (size_t)GG * FF;       // 128 KB
  unsigned short* whhEp = ws + off; off += (size_t)GG * HH;       // 512 KB
  unsigned short* wihDp = ws + off; off += (size_t)GG * FF;
  unsigned short* whhDp = ws + off; off += (size_t)GG * HH;
  unsigned short* woutp = ws + off; off += (size_t)FF * HH;       // 32 KB

  const int nx = BB * TT * FF;
  k_cvt_bf16<<<4096, 256, 0, stream>>>(ts, xbf, nx);
  k_pack<<<(GG * FF) / 256, 256, 0, stream>>>(WihE, wihEp, GG, FF);
  k_pack<<<(GG * HH) / 256, 256, 0, stream>>>(WhhE, whhEp, GG, HH);
  k_pack<<<(GG * FF) / 256, 256, 0, stream>>>(WihD, wihDp, GG, FF);
  k_pack<<<(GG * HH) / 256, 256, 0, stream>>>(WhhD, whhDp, GG, HH);
  k_pack<<<(FF * HH) / 256, 256, 0, stream>>>(Wout, woutp, FF, HH);

  lstm_ed_main<<<BB / 16, 256, 0, stream>>>(xbf, wihEp, whhEp, wihDp, whhDp,
                                            woutp, bE, bD, bout, out);
}
